// PointAttentionBlock_45449343926649
// MI455X (gfx1250) — matrix operand order repaired and checked by reading, not compiled, not run on hardware
//
#include <hip/hip_runtime.h>

// ---------------------------------------------------------------------------
// Problem constants (B=4, N=2048, C=512, H=8, frames=2)
// ---------------------------------------------------------------------------
#define CDIM   512
#define HEADS  8
#define HDIM   64
#define BEFF   2
#define NEFF_  4096          // frames * N
#define MTOT   8192          // BEFF * NEFF_
#define QKVN   1536          // 3*C

typedef __attribute__((ext_vector_type(16))) _Float16 v16h;
typedef __attribute__((ext_vector_type(8)))  _Float16 v8h;
typedef __attribute__((ext_vector_type(8)))  float    v8f;

#if defined(__has_builtin)
#if __has_builtin(__builtin_amdgcn_global_load_async_to_lds_b128)
#define USE_ASYNC_LDS 1
#endif
#endif

#if defined(USE_ASYNC_LDS)
// builtin expects pointers to a 128-bit int vector in AS(1)/AS(3)
typedef __attribute__((__vector_size__(4 * sizeof(int)))) int vs4i;
typedef __attribute__((address_space(1))) vs4i gvs4i;
typedef __attribute__((address_space(3))) vs4i lvs4i;
#endif

union AFrag { v16h full; v8h h2[2]; };

__device__ __forceinline__ v8f wmma_f16(v16h a, v16h b, v8f c) {
  // D = A(16x32) x B(32x16) + C, f32 accumulate
  return __builtin_amdgcn_wmma_f32_16x16x32_f16(
      /*neg_a=*/false, a, /*neg_b=*/false, b,
      /*c_mod=*/(short)0, c, /*reuse_a=*/false, /*reuse_b=*/false);
}

// A-matrix 16x32 fragment: lane holds row (lane&15); halves [k0+16g..+7] in
// v0..3 and [k0+16g+8..+7] in v4..7  (g = lane>>4).  Two 16B loads.
__device__ __forceinline__ v16h load_afrag(const _Float16* __restrict__ p,
                                           int lda, int m0, int k0, int r, int g) {
  AFrag a;
  const _Float16* base = p + (size_t)(m0 + r) * lda + k0 + 16 * g;
  a.h2[0] = *(const v8h*)base;
  a.h2[1] = *(const v8h*)(base + 8);
  return a.full;
}

// B-matrix 32x16 fragment: lane holds column (lane&15); 16 contiguous K
// values starting at k0+16g.  Sources row-major "rows = columns of B"
// storage (weights [out][in], K [tok][hd], V transposed [hd][tok]).
__device__ __forceinline__ v16h load_bfrag(const _Float16* __restrict__ p,
                                           int ldb, int k0, int n0, int r, int g) {
  return *(const v16h*)(p + (size_t)(n0 + r) * ldb + k0 + 16 * g);
}

// 16B global -> LDS copy: async DMA (ASYNCcnt) when available, else bounce.
__device__ __forceinline__ void copy16_g2l(const _Float16* g, _Float16* l) {
#if defined(USE_ASYNC_LDS)
  __builtin_amdgcn_global_load_async_to_lds_b128(
      (gvs4i*)(void*)g, (lvs4i*)(void*)l, 0, 0);
#else
  *(v8h*)l = *(const v8h*)g;
#endif
}

__device__ __forceinline__ void wait_async_all() {
#if defined(USE_ASYNC_LDS)
#if __has_builtin(__builtin_amdgcn_s_wait_asynccnt)
  __builtin_amdgcn_s_wait_asynccnt(0);
#else
  asm volatile("s_wait_asynccnt 0" ::: "memory");
#endif
#endif
}

// ---------------------------------------------------------------------------
// 1) LayerNorm -> f16   (one block per row of 512)
// ---------------------------------------------------------------------------
__global__ __launch_bounds__(256)
void ln_f16_kernel(const float* __restrict__ x, const float* __restrict__ w,
                   const float* __restrict__ b, _Float16* __restrict__ xn) {
  __shared__ float red[256];
  const int row = blockIdx.x, t = threadIdx.x;
  const float* xr = x + (size_t)row * CDIM;
  float a0 = xr[t], a1 = xr[t + 256];
  red[t] = a0 + a1; __syncthreads();
  #pragma unroll
  for (int off = 128; off > 0; off >>= 1) { if (t < off) red[t] += red[t + off]; __syncthreads(); }
  const float mu = red[0] * (1.0f / CDIM);
  __syncthreads();
  const float d0 = a0 - mu, d1 = a1 - mu;
  red[t] = d0 * d0 + d1 * d1; __syncthreads();
  #pragma unroll
  for (int off = 128; off > 0; off >>= 1) { if (t < off) red[t] += red[t + off]; __syncthreads(); }
  const float rstd = rsqrtf(red[0] * (1.0f / CDIM) + 1e-5f);
  _Float16* o = xn + (size_t)row * CDIM;
  o[t]       = (_Float16)(d0 * rstd * w[t]       + b[t]);
  o[t + 256] = (_Float16)(d1 * rstd * w[t + 256] + b[t + 256]);
}

// ---------------------------------------------------------------------------
// 2) Weight f32 -> f16 convert (layouts unchanged: [out][in])
// ---------------------------------------------------------------------------
__global__ __launch_bounds__(256)
void cvt_w_kernel(const float* __restrict__ wqkv, const float* __restrict__ wproj,
                  _Float16* __restrict__ wqkvh, _Float16* __restrict__ wprojh) {
  const int i = blockIdx.x * 256 + threadIdx.x;
  if (i < QKVN * CDIM) wqkvh[i] = (_Float16)wqkv[i];
  if (i < CDIM * CDIM) wprojh[i] = (_Float16)wproj[i];
}

// ---------------------------------------------------------------------------
// 3) QKV GEMM: xn(8192x512) x Wqkv^T(512x1536) + bqkv
//    8 waves/block, 32x32 tile per wave, block tile 128x64.
// ---------------------------------------------------------------------------
__global__ __launch_bounds__(256)
void qkv_gemm_kernel(const _Float16* __restrict__ A, const _Float16* __restrict__ W,
                     const float* __restrict__ bias, _Float16* __restrict__ qh,
                     _Float16* __restrict__ kh, _Float16* __restrict__ vt) {
  const int lane = threadIdx.x & 31, wave = threadIdx.x >> 5;
  const int r = lane & 15, g = lane >> 4;
  const int m0 = blockIdx.y * 128 + (wave >> 1) * 32;
  const int n0 = blockIdx.x * 64  + (wave & 1) * 32;
  v8f acc[2][2] = {};
  for (int k0 = 0; k0 < CDIM; k0 += 32) {
    if (k0 + 32 < CDIM) {
      __builtin_prefetch(A + (size_t)(m0 + r) * CDIM + k0 + 32, 0, 3);
      __builtin_prefetch(W + (size_t)(n0 + r) * CDIM + k0 + 32, 0, 3);
    }
    v16h a0 = load_afrag(A, CDIM, m0,      k0, r, g);
    v16h a1 = load_afrag(A, CDIM, m0 + 16, k0, r, g);
    v16h b0 = load_bfrag(W, CDIM, k0, n0,      r, g);
    v16h b1 = load_bfrag(W, CDIM, k0, n0 + 16, r, g);
    acc[0][0] = wmma_f16(a0, b0, acc[0][0]);
    acc[0][1] = wmma_f16(a0, b1, acc[0][1]);
    acc[1][0] = wmma_f16(a1, b0, acc[1][0]);
    acc[1][1] = wmma_f16(a1, b1, acc[1][1]);
  }
  #pragma unroll
  for (int ti = 0; ti < 2; ti++)
    #pragma unroll
    for (int tj = 0; tj < 2; tj++)
      #pragma unroll
      for (int j = 0; j < 8; j++) {
        const int row = m0 + 16 * ti + j + 8 * g;       // token (0..8191)
        const int col = n0 + 16 * tj + r;               // qkv channel (0..1535)
        const float val = acc[ti][tj][j] + bias[col];
        const int which = col >> 9, hh = (col >> 6) & 7, d = col & 63;
        const int bb = row >> 12, n = row & (NEFF_ - 1);
        const size_t hb = (size_t)(bb * HEADS + hh);
        if (which == 0)      qh[(hb * NEFF_ + n) * HDIM + d] = (_Float16)(val * 0.125f);
        else if (which == 1) kh[(hb * NEFF_ + n) * HDIM + d] = (_Float16)val;
        else                 vt[(hb * HDIM + d) * NEFF_ + n] = (_Float16)val; // transposed
      }
}

// ---------------------------------------------------------------------------
// 4) Flash attention.  Block = 4 waves = 64 query rows of ONE (b,h).
//    K/V chunks (32 keys) are cooperatively staged into double-buffered LDS
//    with async global->LDS DMA, overlapped with WMMA compute on the
//    previous buffer.  Per chunk per wave: 4 WMMAs for S, online f32
//    softmax (half-wave shuffle reductions), P through wave-private LDS
//    into A-frag layout, 4 WMMAs for O += P*V.
// ---------------------------------------------------------------------------
__global__ __launch_bounds__(128)
void attn_kernel(const _Float16* __restrict__ qh, const _Float16* __restrict__ kh,
                 const _Float16* __restrict__ vt, _Float16* __restrict__ ao) {
  __shared__ __align__(32) _Float16 kbuf[2][32 * HDIM];   // 2 x 4KB, [key][hd]
  __shared__ __align__(32) _Float16 vbuf[2][HDIM * 32];   // 2 x 4KB, [hd][key]
  __shared__ __align__(32) _Float16 plds[4][16 * 32];     // per-wave P tile
  const int tid = threadIdx.x;
  const int lane = tid & 31, wave = tid >> 5;
  const int r = lane & 15, g = lane >> 4;
  const int b  = blockIdx.x >> 9;            // / (HEADS * 64)
  const int hh = (blockIdx.x >> 6) & 7;
  const int q0 = (blockIdx.x & 63) * 64 + wave * 16;
  const _Float16* Q = qh + (size_t)(b * HEADS + hh) * NEFF_ * HDIM;
  const _Float16* K = kh + (size_t)(b * HEADS + hh) * NEFF_ * HDIM;
  const _Float16* V = vt + (size_t)(b * HEADS + hh) * HDIM * NEFF_;
  _Float16* pl = plds[wave];

  // cooperative stage of one 32-key chunk: K rows are contiguous (4KB),
  // V (transposed) rows are 64B segments.  256+256 16B copies, 4/thread.
  auto stage_kv = [&](int buf, int kv) {
    #pragma unroll
    for (int c = tid; c < 256; c += 128)
      copy16_g2l(K + (size_t)kv * HDIM + c * 8, &kbuf[buf][c * 8]);
    #pragma unroll
    for (int c = tid; c < 256; c += 128) {
      const int d = c >> 2, s = c & 3;
      copy16_g2l(V + (size_t)d * NEFF_ + kv + s * 8, &vbuf[buf][d * 32 + s * 8]);
    }
  };

  const v16h qa0 = load_afrag(Q, HDIM, q0, 0,  r, g);
  const v16h qa1 = load_afrag(Q, HDIM, q0, 32, r, g);
  v8f o[4] = {};
  float mrow[8], lrow[8];
  #pragma unroll
  for (int j = 0; j < 8; j++) { mrow[j] = -3.0e38f; lrow[j] = 0.0f; }

  stage_kv(0, 0);
  wait_async_all();
  __syncthreads();

  int cur = 0;
  for (int kv = 0; kv < NEFF_; kv += 32) {
    if (kv + 32 < NEFF_) stage_kv(cur ^ 1, kv + 32);   // overlap DMA with compute
    const _Float16* kl = kbuf[cur];
    const _Float16* vl = vbuf[cur];

    // S = (Q*scale) K^T for keys [kv, kv+32), K fragments from LDS
    v16h kb00 = *(const v16h*)(kl + (r)      * HDIM + 0  + 16 * g);
    v16h kb01 = *(const v16h*)(kl + (r)      * HDIM + 32 + 16 * g);
    v16h kb10 = *(const v16h*)(kl + (16 + r) * HDIM + 0  + 16 * g);
    v16h kb11 = *(const v16h*)(kl + (16 + r) * HDIM + 32 + 16 * g);
    v8f s0 = {}, s1 = {};
    s0 = wmma_f16(qa0, kb00, s0);  s0 = wmma_f16(qa1, kb01, s0);
    s1 = wmma_f16(qa0, kb10, s1);  s1 = wmma_f16(qa1, kb11, s1);

    // online softmax: row j lives in half-wave g across lanes [16g, 16g+16)
    float alpha[8];
    #pragma unroll
    for (int j = 0; j < 8; j++) {
      float mx = fmaxf(s0[j], s1[j]);
      #pragma unroll
      for (int off = 1; off < 16; off <<= 1) mx = fmaxf(mx, __shfl_xor(mx, off, 32));
      const float mn = fmaxf(mrow[j], mx);
      const float p0 = __expf(s0[j] - mn);
      const float p1 = __expf(s1[j] - mn);
      float rs = p0 + p1;
      #pragma unroll
      for (int off = 1; off < 16; off <<= 1) rs += __shfl_xor(rs, off, 32);
      alpha[j] = __expf(mrow[j] - mn);
      lrow[j]  = lrow[j] * alpha[j] + rs;
      mrow[j]  = mn;
      pl[(j + 8 * g) * 32 + r]      = (_Float16)p0;   // P row-major 16x32
      pl[(j + 8 * g) * 32 + 16 + r] = (_Float16)p1;
    }
    #pragma unroll
    for (int t = 0; t < 4; t++) {
      v8f tmp = o[t];
      #pragma unroll
      for (int j = 0; j < 8; j++) tmp[j] *= alpha[j];
      o[t] = tmp;
    }

    // O += P(16x32) * V(32x64): P re-read in A-frag layout (wave-private
    // LDS; same-wave DS ops are in-order), V fragments from staged LDS.
    AFrag pa;
    const _Float16* pb = pl + r * 32 + 16 * g;
    pa.h2[0] = *(const v8h*)pb;
    pa.h2[1] = *(const v8h*)(pb + 8);
    #pragma unroll
    for (int t = 0; t < 4; t++) {
      v16h vb = *(const v16h*)(vl + (16 * t + r) * 32 + 16 * g);
      o[t] = wmma_f16(pa.full, vb, o[t]);
    }

    wait_async_all();      // own asyncs for next buffer done ...
    __syncthreads();       // ... and visible block-wide; cur buffer released
    cur ^= 1;
  }

  // normalize and store attention output as f16 [token][C]
  #pragma unroll
  for (int j = 0; j < 8; j++) {
    const float inv = 1.0f / lrow[j];
    const int tok = b * NEFF_ + q0 + j + 8 * g;
    _Float16* dst = ao + (size_t)tok * CDIM + hh * HDIM;
    #pragma unroll
    for (int t = 0; t < 4; t++) dst[16 * t + r] = (_Float16)(o[t][j] * inv);
  }
}

// ---------------------------------------------------------------------------
// 5) Proj GEMM: ao(8192x512) x Wproj^T(512x512) + bproj + residual -> f32 out
// ---------------------------------------------------------------------------
__global__ __launch_bounds__(256)
void proj_gemm_kernel(const _Float16* __restrict__ A, const _Float16* __restrict__ W,
                      const float* __restrict__ bias, const float* __restrict__ resid,
                      float* __restrict__ out) {
  const int lane = threadIdx.x & 31, wave = threadIdx.x >> 5;
  const int r = lane & 15, g = lane >> 4;
  const int m0 = blockIdx.y * 128 + (wave >> 1) * 32;
  const int n0 = blockIdx.x * 64  + (wave & 1) * 32;
  v8f acc[2][2] = {};
  for (int k0 = 0; k0 < CDIM; k0 += 32) {
    if (k0 + 32 < CDIM) {
      __builtin_prefetch(A + (size_t)(m0 + r) * CDIM + k0 + 32, 0, 3);
      __builtin_prefetch(W + (size_t)(n0 + r) * CDIM + k0 + 32, 0, 3);
    }
    v16h a0 = load_afrag(A, CDIM, m0,      k0, r, g);
    v16h a1 = load_afrag(A, CDIM, m0 + 16, k0, r, g);
    v16h b0 = load_bfrag(W, CDIM, k0, n0,      r, g);
    v16h b1 = load_bfrag(W, CDIM, k0, n0 + 16, r, g);
    acc[0][0] = wmma_f16(a0, b0, acc[0][0]);
    acc[0][1] = wmma_f16(a0, b1, acc[0][1]);
    acc[1][0] = wmma_f16(a1, b0, acc[1][0]);
    acc[1][1] = wmma_f16(a1, b1, acc[1][1]);
  }
  #pragma unroll
  for (int ti = 0; ti < 2; ti++)
    #pragma unroll
    for (int tj = 0; tj < 2; tj++)
      #pragma unroll
      for (int j = 0; j < 8; j++) {
        const int row = m0 + 16 * ti + j + 8 * g;
        const int col = n0 + 16 * tj + r;
        const size_t idx = (size_t)row * CDIM + col;
        out[idx] = resid[idx] + acc[ti][tj][j] + bias[col];
      }
}

// ---------------------------------------------------------------------------
// Launch
// ---------------------------------------------------------------------------
extern "C" void kernel_launch(void* const* d_in, const int* in_sizes, int n_in,
                              void* d_out, int out_size, void* d_ws, size_t ws_size,
                              hipStream_t stream) {
  (void)in_sizes; (void)n_in; (void)out_size; (void)ws_size;
  const float* x     = (const float*)d_in[0];
  const float* ln_w  = (const float*)d_in[1];
  const float* ln_b  = (const float*)d_in[2];
  const float* Wqkv  = (const float*)d_in[3];
  const float* bqkv  = (const float*)d_in[4];
  const float* Wproj = (const float*)d_in[5];
  const float* bproj = (const float*)d_in[6];
  float* out = (float*)d_out;

  // workspace carve-up (256B aligned)
  char* ws = (char*)d_ws; size_t off = 0;
  auto take = [&](size_t bytes) { char* p = ws + off; off += (bytes + 255) & ~(size_t)255; return p; };
  _Float16* xn     = (_Float16*)take((size_t)MTOT * CDIM * 2);      // 8 MB
  _Float16* wqkvh  = (_Float16*)take((size_t)QKVN * CDIM * 2);      // 1.5 MB
  _Float16* wprojh = (_Float16*)take((size_t)CDIM * CDIM * 2);      // 0.5 MB
  _Float16* qh     = (_Float16*)take((size_t)BEFF * HEADS * NEFF_ * HDIM * 2); // 8 MB
  _Float16* kh     = (_Float16*)take((size_t)BEFF * HEADS * NEFF_ * HDIM * 2); // 8 MB
  _Float16* vt     = (_Float16*)take((size_t)BEFF * HEADS * HDIM * NEFF_ * 2); // 8 MB
  _Float16* ao     = (_Float16*)take((size_t)MTOT * CDIM * 2);      // 8 MB

  cvt_w_kernel<<<(QKVN * CDIM) / 256, 256, 0, stream>>>(Wqkv, Wproj, wqkvh, wprojh);
  ln_f16_kernel<<<MTOT, 256, 0, stream>>>(x, ln_w, ln_b, xn);
  qkv_gemm_kernel<<<dim3(QKVN / 64, MTOT / 128), 256, 0, stream>>>(xn, wqkvh, bqkv, qh, kh, vt);
  attn_kernel<<<BEFF * HEADS * (NEFF_ / 64), 128, 0, stream>>>(qh, kh, vt, ao);
  proj_gemm_kernel<<<dim3(CDIM / 64, MTOT / 128), 256, 0, stream>>>(ao, wprojh, bproj, x, out);
}
